// GRUModel_1202590842921
// MI455X (gfx1250) — matrix operand order repaired and checked by reading, not compiled, not run on hardware
//
#include <hip/hip_runtime.h>
#include <hip/hip_bf16.h>
#include <math.h>

// ---------------------------------------------------------------------------
// GRU: B=64, T=512, IN=256, H=1024, OUT=10
// Strategy: persistent bf16-WMMA scan kernel, fused x/h GEMM, gate-local
// elementwise, device-wide atomic barrier per timestep, h kept in registers.
// ---------------------------------------------------------------------------

typedef __attribute__((ext_vector_type(16))) __bf16 v16bf;
typedef __attribute__((ext_vector_type(8)))  float  v8f;

// ---- workspace layout (bytes) ----
// Tile = 32 lanes x 16 bf16 = 1024 B; per-lane chunk = 32 B contiguous.
static constexpr size_t OFF_XA   = 0;                       // [T][4 mt][8 kt] tiles   : 16,777,216
static constexpr size_t OFF_WR   = 16777216;                // [64 nt][40 kt]          :  2,621,440
static constexpr size_t OFF_WI   = 19398656;                // [64 nt][40 kt]          :  2,621,440
static constexpr size_t OFF_WXN  = 22020096;                // [64 nt][ 8 kt]          :    524,288
static constexpr size_t OFF_WHN  = 22544384;                // [64 nt][32 kt]          :  2,097,152
static constexpr size_t OFF_HA   = 24641536;                // 2 bufs x [4 mt][32 kt]  :    262,144
static constexpr size_t OFF_HF   = 24903680;                // h_last f32 [64][1024]   :    262,144
static constexpr size_t OFF_BIAS = 25165824;                // br|bi|bxn|bhn 4x1024 f32:     16,384
static constexpr size_t OFF_BAR  = 25182208;                // barrier counter

static __device__ inline v16bf ldtile(const char* p) {
    return *(const v16bf*)p;            // 32B -> 2x global_load_b128
}
static __device__ inline v8f wmma_bf16(v16bf a, v16bf b, v8f c) {
    return __builtin_amdgcn_wmma_f32_16x16x32_bf16(false, a, false, b, (short)0, c, false, false);
}
static __device__ inline float fsigmoid(float x) { return 1.0f / (1.0f + __expf(-x)); }
static __device__ inline float ftanh(float x) {
    float e = __expf(-2.0f * x);
    return (1.0f - e) / (1.0f + e);
}

// ---------------------------------------------------------------------------
// Init: zero hA double-buffer + barrier counter, build combined biases.
// ---------------------------------------------------------------------------
__global__ void gru_init(char* __restrict__ ws, const float* __restrict__ bx,
                         const float* __restrict__ bh) {
    int id = blockIdx.x * 256 + threadIdx.x;
    if (id < 65536) ((unsigned*)(ws + OFF_HA))[id] = 0u;     // both h_A buffers = 0
    int j = id - 65536;
    if (j >= 0 && j < 1024) {
        float* bias = (float*)(ws + OFF_BIAS);
        bias[j]        = bx[j]        + bh[j];               // br
        bias[1024 + j] = bx[1024 + j] + bh[1024 + j];        // bi
        bias[2048 + j] = bx[2048 + j];                       // bxn
        bias[3072 + j] = bh[2048 + j];                       // bhn
    }
    if (id == 0) *(unsigned*)(ws + OFF_BAR) = 0u;
}

// ---------------------------------------------------------------------------
// Pack weights into WMMA B-layout (32x16 bf16 tiles).
// B element (k,n): lane = (n&15) | (((k>>3)&1)<<4); slot s = ((k>>4)<<3)|(k&7) (k within tile).
// ---------------------------------------------------------------------------
__global__ void gru_pack_w(char* __restrict__ ws, const float* __restrict__ Wx,
                           const float* __restrict__ Wh) {
    int tid = blockIdx.x * 256 + threadIdx.x;   // 245760 threads
    int lane = tid & 31, tile = tid >> 5;
    int nl  = lane & 15;
    int khi = lane >> 4;
    char* dst; int row, kt; int mode;           // 0: fused r/i, 1: xn, 2: hn
    if (tile < 2560)      { int tt = tile;        int nt = tt / 40; kt = tt % 40; row =        nt * 16 + nl; dst = ws + OFF_WR  + ((size_t)tt * 32 + lane) * 32; mode = 0; }
    else if (tile < 5120) { int tt = tile - 2560; int nt = tt / 40; kt = tt % 40; row = 1024 + nt * 16 + nl; dst = ws + OFF_WI  + ((size_t)tt * 32 + lane) * 32; mode = 0; }
    else if (tile < 5632) { int tt = tile - 5120; int nt = tt /  8; kt = tt %  8; row = 2048 + nt * 16 + nl; dst = ws + OFF_WXN + ((size_t)tt * 32 + lane) * 32; mode = 1; }
    else                  { int tt = tile - 5632; int nt = tt / 32; kt = tt % 32; row = 2048 + nt * 16 + nl; dst = ws + OFF_WHN + ((size_t)tt * 32 + lane) * 32; mode = 2; }
    __bf16* d = (__bf16*)dst;
    for (int s = 0; s < 16; ++s) {
        int kg = kt * 32 + ((s >> 3) << 4) + (khi << 3) + (s & 7);
        float v;
        if (mode == 2)      v = Wh[(size_t)row * 1024 + kg];
        else if (mode == 1) v = Wx[(size_t)row * 256 + kg];
        else                v = (kg < 256) ? Wx[(size_t)row * 256 + kg]
                                           : Wh[(size_t)row * 1024 + (kg - 256)];
        d[s] = (__bf16)v;
    }
}

// ---------------------------------------------------------------------------
// Pack x into WMMA A-layout tiles for every timestep.
// A element (m,k): lane = (m&15) | (((k>>3)&1)<<4); slot s = ((k>>4)<<3)|(k&7).
// ---------------------------------------------------------------------------
__global__ void gru_pack_x(char* __restrict__ ws, const float* __restrict__ x) {
    int tid = blockIdx.x * 256 + threadIdx.x;   // 524288 threads
    int lane = tid & 31;
    int rest = tid >> 5;
    int kt = rest & 7, mt = (rest >> 3) & 3, t = rest >> 5;
    int ml = lane & 15;
    int lhi = lane >> 4;
    int bb = mt * 16 + ml;
    __bf16* d = (__bf16*)(ws + OFF_XA + ((size_t)(((t * 4 + mt) * 8) + kt) * 32 + lane) * 32);
    for (int s = 0; s < 16; ++s) {
        int kl = ((s >> 3) << 4) + (lhi << 3) + (s & 7);
        int k = kt * 32 + kl;
        d[s] = (__bf16)x[((size_t)bb * 512 + t) * 256 + k];
    }
}

// ---------------------------------------------------------------------------
// Persistent scan. 64 WGs x 8 waves. WG g owns j in [16g,16g+16).
// wave = (kh,mt): mt = M tile of batch, kh = K-split half (balanced 60 WMMA).
// Per step: GEMM accum -> LDS K-reduce -> wave-local GRU update (h in regs)
//           -> scatter h' (bf16, A-layout) -> device barrier.
// ---------------------------------------------------------------------------
__global__ __launch_bounds__(256) void gru_scan(char* __restrict__ ws) {
    const int g = blockIdx.x;
    const int tid = threadIdx.x;
    const int lane = tid & 31;
    const int wave = tid >> 5;
    const int mt = wave & 3;
    const int kh = wave >> 2;

    const char* xA  = ws + OFF_XA;
    const char* Wr  = ws + OFF_WR;
    const char* Wi  = ws + OFF_WI;
    const char* Wxn = ws + OFF_WXN;
    const char* Whn = ws + OFF_WHN;
    char* hA = ws + OFF_HA;
    float* hf = (float*)(ws + OFF_HF);
    const float* bias = (const float*)(ws + OFF_BIAS);
    unsigned* bar = (unsigned*)(ws + OFF_BAR);

    __shared__ float red[4][3][256];   // [mt][r,i,hn][v*32+lane] = 12 KB

    const int jj = lane & 15;
    const int jglob = g * 16 + jj;
    const float brl  = bias[jglob];
    const float bil  = bias[1024 + jglob];
    const float bxnl = bias[2048 + jglob];
    const float bhnl = bias[3072 + jglob];
    const int ktj = jglob >> 5;
    const int klj = jglob & 31;
    const int sA  = ((klj >> 4) << 3) | (klj & 7);

    float hreg[8];
#pragma unroll
    for (int v = 0; v < 8; ++v) hreg[v] = 0.0f;

    for (int t = 0; t < 512; ++t) {
        const int buf = t & 1;
        const char* hAr = hA + (size_t)buf * 131072;
        char* hAw = hA + (size_t)(buf ^ 1) * 131072;

        v8f accr = {}, acci = {}, accxn = {}, acchn = {};

        if (kh == 0) {
            // x part: fused kt 0..7 (r, i, xn)
#pragma unroll
            for (int kt = 0; kt < 8; ++kt) {
                v16bf a  = ldtile(xA + ((size_t)(((t * 4 + mt) * 8) + kt) * 32 + lane) * 32);
                v16bf wr = ldtile(Wr  + ((size_t)(g * 40 + kt) * 32 + lane) * 32);
                accr = wmma_bf16(a, wr, accr);
                v16bf wi = ldtile(Wi  + ((size_t)(g * 40 + kt) * 32 + lane) * 32);
                acci = wmma_bf16(a, wi, acci);
                v16bf wx = ldtile(Wxn + ((size_t)(g * 8 + kt) * 32 + lane) * 32);
                accxn = wmma_bf16(a, wx, accxn);
            }
            // h part: hkt 0..11 (r, i, hn)
#pragma unroll
            for (int hkt = 0; hkt < 12; ++hkt) {
                v16bf a  = ldtile(hAr + ((size_t)(mt * 32 + hkt) * 32 + lane) * 32);
                v16bf wr = ldtile(Wr  + ((size_t)(g * 40 + 8 + hkt) * 32 + lane) * 32);
                accr = wmma_bf16(a, wr, accr);
                v16bf wi = ldtile(Wi  + ((size_t)(g * 40 + 8 + hkt) * 32 + lane) * 32);
                acci = wmma_bf16(a, wi, acci);
                v16bf wh = ldtile(Whn + ((size_t)(g * 32 + hkt) * 32 + lane) * 32);
                acchn = wmma_bf16(a, wh, acchn);
            }
        } else {
            // h part: hkt 12..31 (r, i, hn)
#pragma unroll
            for (int hkt = 12; hkt < 32; ++hkt) {
                v16bf a  = ldtile(hAr + ((size_t)(mt * 32 + hkt) * 32 + lane) * 32);
                v16bf wr = ldtile(Wr  + ((size_t)(g * 40 + 8 + hkt) * 32 + lane) * 32);
                accr = wmma_bf16(a, wr, accr);
                v16bf wi = ldtile(Wi  + ((size_t)(g * 40 + 8 + hkt) * 32 + lane) * 32);
                acci = wmma_bf16(a, wi, acci);
                v16bf wh = ldtile(Whn + ((size_t)(g * 32 + hkt) * 32 + lane) * 32);
                acchn = wmma_bf16(a, wh, acchn);
            }
            // stage partials for K-reduction
#pragma unroll
            for (int v = 0; v < 8; ++v) {
                red[mt][0][v * 32 + lane] = accr[v];
                red[mt][1][v * 32 + lane] = acci[v];
                red[mt][2][v * 32 + lane] = acchn[v];
            }
        }
        __syncthreads();

        if (kh == 0) {
#pragma unroll
            for (int v = 0; v < 8; ++v) {
                float r  = accr[v]  + red[mt][0][v * 32 + lane] + brl;
                float ig = acci[v]  + red[mt][1][v * 32 + lane] + bil;
                float hn = acchn[v] + red[mt][2][v * 32 + lane] + bhnl;
                float xn = accxn[v] + bxnl;
                r  = fsigmoid(r);
                ig = fsigmoid(ig);
                float nn = ftanh(xn + r * hn);
                float hy = nn + ig * (hreg[v] - nn);
                hreg[v] = hy;
                // scatter into bf16 A-layout for next step's GEMM
                int bb = v + ((lane >> 4) << 3);                 // b & 15
                int lane_a = bb | (((klj >> 3) & 1) << 4);
                *(__bf16*)(hAw + ((size_t)(mt * 32 + ktj) * 32 + lane_a) * 32 + sA * 2) = (__bf16)hy;
            }
        }

        // ---- device-wide barrier ----
        __threadfence();
        __syncthreads();
        if (tid == 0) {
            __hip_atomic_fetch_add(bar, 1u, __ATOMIC_RELEASE, __HIP_MEMORY_SCOPE_AGENT);
            const unsigned target = 64u * (unsigned)(t + 1);
            while (__hip_atomic_load(bar, __ATOMIC_RELAXED, __HIP_MEMORY_SCOPE_AGENT) < target) {
                __builtin_amdgcn_s_sleep(1);
            }
        }
        __syncthreads();
        __builtin_amdgcn_fence(__ATOMIC_ACQUIRE, "agent");   // refetch h panel past WGP$
    }

    if (kh == 0) {
#pragma unroll
        for (int v = 0; v < 8; ++v) {
            int bb = mt * 16 + v + ((lane >> 4) << 3);
            hf[(size_t)bb * 1024 + jglob] = hreg[v];
        }
    }
}

// ---------------------------------------------------------------------------
// Final FC: out[b][o] = h_last[b] . Wfc[o] + bfc[o]   (64x10, trivial)
// ---------------------------------------------------------------------------
__global__ void gru_fc(const char* __restrict__ ws, const float* __restrict__ Wfc,
                       const float* __restrict__ bfc, float* __restrict__ out) {
    int tid = blockIdx.x * 256 + threadIdx.x;
    if (tid >= 640) return;
    int b = tid / 10, o = tid % 10;
    const float* hf = (const float*)(ws + OFF_HF);
    float acc = bfc[o];
    for (int j = 0; j < 1024; ++j)
        acc += hf[(size_t)b * 1024 + j] * Wfc[(size_t)o * 1024 + j];
    out[b * 10 + o] = acc;
}

// ---------------------------------------------------------------------------
extern "C" void kernel_launch(void* const* d_in, const int* in_sizes, int n_in,
                              void* d_out, int out_size, void* d_ws, size_t ws_size,
                              hipStream_t stream) {
    (void)in_sizes; (void)n_in; (void)out_size; (void)ws_size;
    const float* x   = (const float*)d_in[0];
    const float* Wx  = (const float*)d_in[1];
    const float* bx  = (const float*)d_in[2];
    const float* Wh  = (const float*)d_in[3];
    const float* bh  = (const float*)d_in[4];
    const float* Wfc = (const float*)d_in[5];
    const float* bfc = (const float*)d_in[6];
    float* out = (float*)d_out;
    char* ws = (char*)d_ws;

    gru_init  <<<260,  256, 0, stream>>>(ws, bx, bh);
    gru_pack_w<<<960,  256, 0, stream>>>(ws, Wx, Wh);
    gru_pack_x<<<2048, 256, 0, stream>>>(ws, x);
    gru_scan  <<<64,   256, 0, stream>>>(ws);
    gru_fc    <<<3,    256, 0, stream>>>(ws, Wfc, bfc, out);
}
